// NoiseCombSource_15341623181412
// MI455X (gfx1250) — matrix operand order repaired and checked by reading, not compiled, not run on hardware
//
#include <hip/hip_runtime.h>
#include <hip/hip_bf16.h>

typedef __attribute__((ext_vector_type(16))) _Float16 v16h;
typedef __attribute__((ext_vector_type(8)))  float    v8f;

#define TT 2048          // frames
#define CIN 128          // channels
#define NBATCH 8
#define LSAMP 1048576    // T * HOP
#define NB 256           // scan blocks per batch
#define BS 4096          // samples per scan block (NB*BS = LSAMP)
#define SR 44100.0

// ---------------------------------------------------------------------------
// Conv1d as WMMA GEMM: one wave computes a 16(oc) x 16(t) tile.
// K dimension = CIN*TAPS with kappa = tap*CIN + ic, so every 32-wide K slice
// has a constant tap and a contiguous 32-channel run.
// Fragments are built via direct ext-vector subscripts (register-resident,
// no scratch).
// ---------------------------------------------------------------------------
template <int TAPS, int PAD, int RELU>
__global__ __launch_bounds__(32)
void conv_wmma(const float* __restrict__ x, const float* __restrict__ w,
               const float* __restrict__ bias, float* __restrict__ y) {
  const int lane = threadIdx.x & 31;
  const int m16  = lane & 15;   // A row / B column / C column
  const int hi   = lane >> 4;

  const int bt  = blockIdx.x;          // b*128 + t-tile
  const int b   = bt >> 7;
  const int t0  = (bt & 127) << 4;
  const int oc0 = blockIdx.y << 4;

  // C/D layout: VGPR j holds row M = j + 8*hi, column N = lane&15
  v8f acc;
#pragma unroll
  for (int j = 0; j < 8; ++j) acc[j] = bias[oc0 + j + 8 * hi];

  const float* xb = x + (size_t)b * CIN * TT;
  const int nslices = (CIN * TAPS) >> 5;

  for (int s = 0; s < nslices; ++s) {
    const int kt  = (s * 32) / CIN;        // tap index (constant in slice)
    const int icb = (s * 32) % CIN;        // channel base (32-aligned)

    v16h Af, Bf;

    // A fragment (weights 16x32 f16): lane row m16; VGPR j holds
    // K = hi*8+2j (j<4) else 16+hi*8+2(j-4), and K+1.
    const float* wr = w + ((size_t)(oc0 + m16) * CIN) * TAPS + kt;
#pragma unroll
    for (int j = 0; j < 8; ++j) {
      const int k0  = (j < 4) ? (hi * 8 + 2 * j) : (16 + hi * 8 + 2 * (j - 4));
      const int ic0 = icb + k0;
      Af[2 * j]     = (_Float16)wr[(size_t)ic0 * TAPS];
      Af[2 * j + 1] = (_Float16)wr[(size_t)(ic0 + 1) * TAPS];
    }

    // B fragment (patch 32x16 f16): column n = lane&15;
    // VGPR j holds K = hi*8+2j (j<4) else 16+hi*8+2(j-4), and K+1.
    const int tcol = t0 + m16 + kt - PAD;
    const bool inb = (tcol >= 0) && (tcol < TT);
    const int tc   = inb ? tcol : 0;       // clamp, then mask (keeps EXEC full)
#pragma unroll
    for (int j = 0; j < 8; ++j) {
      const int ic0 = icb + ((j < 4) ? (hi * 8 + 2 * j) : (16 + hi * 8 + 2 * (j - 4)));
      float v0 = xb[(size_t)ic0 * TT + tc];
      float v1 = xb[(size_t)(ic0 + 1) * TT + tc];
      if (!inb) { v0 = 0.f; v1 = 0.f; }
      Bf[2 * j]     = (_Float16)v0;
      Bf[2 * j + 1] = (_Float16)v1;
    }

    acc = __builtin_amdgcn_wmma_f32_16x16x32_f16(false, Af, false, Bf,
                                                 (short)0, acc, false, false);
  }

  float* yb = y + (size_t)b * CIN * TT;
#pragma unroll
  for (int j = 0; j < 8; ++j) {
    const int oc = oc0 + j + 8 * hi;
    float v = acc[j];
    if (RELU) v = (v >= 0.f) ? v : 0.2f * v;
    yb[(size_t)oc * TT + t0 + m16] = v;
  }
}

// ---------------------------------------------------------------------------
// Conv3: 2 output channels, taps=3, pad=1, clip to [0,1] -> mags [B,2,T]
// ---------------------------------------------------------------------------
__global__ void conv3_mags(const float* __restrict__ h, const float* __restrict__ w3,
                           const float* __restrict__ b3, float* __restrict__ mags) {
  const int idx = blockIdx.x * blockDim.x + threadIdx.x;   // B*2*T
  if (idx >= NBATCH * 2 * TT) return;
  const int t  = idx % TT;
  const int ch = (idx / TT) % 2;
  const int b  = idx / (2 * TT);
  const float* hb = h + (size_t)b * CIN * TT;
  float acc = b3[ch];
  for (int ic = 0; ic < CIN; ++ic) {
    const float* wr = w3 + ((size_t)ch * CIN + ic) * 3;
    const float* hr = hb + (size_t)ic * TT;
    if (t - 1 >= 0) acc += wr[0] * hr[t - 1];
    acc += wr[1] * hr[t];
    if (t + 1 < TT) acc += wr[2] * hr[t + 1];
  }
  mags[idx] = fminf(fmaxf(acc, 0.f), 1.f);
}

// ---------------------------------------------------------------------------
// align_corners lerp of a [T] frame track at sample n (f32 weights like ref)
// ---------------------------------------------------------------------------
__device__ __forceinline__ float lerp_frame(const float* __restrict__ fr, int n) {
  const double pos = (double)n * (2047.0 / 1048575.0);
  const int i0 = (int)pos;
  const int i1 = (i0 + 1 < TT) ? i0 + 1 : TT - 1;
  const float wf = (float)(pos - (double)i0);
  return fr[i0] * (1.f - wf) + fr[i1] * wf;
}

// ---------------------------------------------------------------------------
// Phase scan, pass 1: per-block (4096-sample) sums of f0/SR in double
// ---------------------------------------------------------------------------
__global__ void p1_blocksums(const float* __restrict__ f0, double* __restrict__ bsum) {
  const int blk = blockIdx.x;              // b*NB + j
  const int b = blk / NB, j = blk % NB;
  const float* f0b = f0 + (size_t)b * TT;
  const int n0 = j * BS + threadIdx.x * 16;
  double s = 0.0;
  for (int i = 0; i < 16; ++i) s += (double)lerp_frame(f0b, n0 + i) / SR;
  __shared__ double sh[256];
  sh[threadIdx.x] = s;
  __syncthreads();
  for (int off = 128; off > 0; off >>= 1) {
    if (threadIdx.x < off) sh[threadIdx.x] += sh[threadIdx.x + off];
    __syncthreads();
  }
  if (threadIdx.x == 0) bsum[blk] = sh[0];
}

// Pass 2: per-batch exclusive scan of the 256 block sums
__global__ void p2_scan(const double* __restrict__ bsum, double* __restrict__ boff) {
  const int b = blockIdx.x, tid = threadIdx.x;
  __shared__ double sh[256];
  const double v = bsum[b * NB + tid];
  sh[tid] = v;
  __syncthreads();
  for (int off = 1; off < 256; off <<= 1) {
    const double t = (tid >= off) ? sh[tid - off] : 0.0;
    __syncthreads();
    sh[tid] += t;
    __syncthreads();
  }
  boff[b * NB + tid] = sh[tid] - v;        // exclusive prefix
}

// ---------------------------------------------------------------------------
// Pass 3 (roofline kernel): phase -> sinc comb, mags lerp, clips, 3 outputs
// ---------------------------------------------------------------------------
__global__ void p3_final(const float* __restrict__ f0, const float* __restrict__ noise,
                         const float* __restrict__ mags, const double* __restrict__ boff,
                         float* __restrict__ out) {
  const size_t BL = (size_t)NBATCH * LSAMP;
  const int blk = blockIdx.x;
  const int b = blk / NB, j = blk % NB;
  const int tid = threadIdx.x;
  const float* f0b = f0 + (size_t)b * TT;
  const int n0 = j * BS + tid * 16;

  float fv[16];
  double run = 0.0;
#pragma unroll
  for (int i = 0; i < 16; ++i) {
    fv[i] = lerp_frame(f0b, n0 + i);
    run += (double)fv[i] / SR;
  }

  __shared__ double sh[256];
  sh[tid] = run;
  __syncthreads();
  const double mine = run;
  for (int off = 1; off < 256; off <<= 1) {
    const double t = (tid >= off) ? sh[tid - off] : 0.0;
    __syncthreads();
    sh[tid] += t;
    __syncthreads();
  }
  double acc = boff[blk] + (sh[tid] - mine);   // exclusive base for my 16 samples

  const float* mg = mags + (size_t)b * 2 * TT;
  float* sig  = out + (size_t)b * LSAMP;
  float* comb = out + BL + (size_t)b * LSAMP;
  float* nsig = out + 2 * BL + (size_t)b * LSAMP;
  const float* nz = noise + (size_t)b * LSAMP;

#pragma unroll
  for (int i = 0; i < 16; ++i) {
    const int n = n0 + i;
    const float f = fv[i];
    acc += (double)f / SR;                     // inclusive phase (f64, like ref)
    const double ph = acc - rint(acc);         // wrap, round-half-even
    const float phf = (float)ph;
    const float x = 44100.0f * phf / (f + 1e-8f);
    float ct;
    if (x == 0.f) ct = 1.f;
    else { const float px = 3.14159265358979323846f * x; ct = sinf(px) / px; }

    // mags lerp at sample n
    const double pos = (double)n * (2047.0 / 1048575.0);
    const int i0 = (int)pos;
    const int i1 = (i0 + 1 < TT) ? i0 + 1 : TT - 1;
    const float wf = (float)(pos - (double)i0);
    const float m0 = mg[i0] * (1.f - wf) + mg[i1] * wf;
    const float m1 = mg[TT + i0] * (1.f - wf) + mg[TT + i1] * wf;

    const float ns = fminf(fmaxf(nz[n] * m0, -1.f), 1.f);
    const float cs = fminf(fmaxf(ct * m1, -1.f), 1.f);
    const float sg = fminf(fmaxf(ns + cs, -1.f), 1.f);
    sig[n] = sg; comb[n] = cs; nsig[n] = ns;
  }
}

// ---------------------------------------------------------------------------
extern "C" void kernel_launch(void* const* d_in, const int* in_sizes, int n_in,
                              void* d_out, int out_size, void* d_ws, size_t ws_size,
                              hipStream_t stream) {
  (void)in_sizes; (void)n_in; (void)out_size; (void)ws_size;
  const float* mel   = (const float*)d_in[0];
  const float* f0    = (const float*)d_in[1];
  const float* noise = (const float*)d_in[2];
  const float* w1    = (const float*)d_in[3];
  const float* b1    = (const float*)d_in[4];
  const float* w2    = (const float*)d_in[5];
  const float* b2    = (const float*)d_in[6];
  const float* w3    = (const float*)d_in[7];
  const float* b3    = (const float*)d_in[8];
  float* out = (float*)d_out;

  char* ws = (char*)d_ws;
  float*  h1   = (float*)ws;                                  // 8 MiB
  float*  h2   = (float*)(ws + (size_t)8 * 1024 * 1024);      // 8 MiB
  float*  mags = (float*)(ws + (size_t)16 * 1024 * 1024);     // 128 KiB
  double* bsum = (double*)(ws + (size_t)16 * 1024 * 1024 + 131072);
  double* boff = bsum + NBATCH * NB;

  // magnitude network: WMMA convs
  conv_wmma<7, 3, 1><<<dim3(NBATCH * (TT / 16), CIN / 16), 32, 0, stream>>>(mel, w1, b1, h1);
  conv_wmma<3, 1, 1><<<dim3(NBATCH * (TT / 16), CIN / 16), 32, 0, stream>>>(h1, w2, b2, h2);
  conv3_mags<<<(NBATCH * 2 * TT + 255) / 256, 256, 0, stream>>>(h2, w3, b3, mags);

  // f64 phase scan + sample synthesis
  p1_blocksums<<<NBATCH * NB, 256, 0, stream>>>(f0, bsum);
  p2_scan<<<NBATCH, 256, 0, stream>>>(bsum, boff);
  p3_final<<<NBATCH * NB, 256, 0, stream>>>(f0, noise, mags, boff, out);
}
